// FieldlineGraphForecaster_74466142978643
// MI455X (gfx1250) — compile-verified, matrix operand order checked by reading, not run on hardware
//
#include <hip/hip_runtime.h>
#include <hip/hip_bf16.h>
#include <math.h>

// ---------------------------------------------------------------------------
// FieldlineGraphForecaster for MI455X (gfx1250, wave32, WMMA).
// Edge/node MLP GEMMs on v_wmma_f32_16x16x32_bf16 (f32 accumulate); residual
// stream h kept in f32; bf16 mirror hbf feeds the matrix pipes. 64 rows per
// block (4 M-tiles) so each B fragment load feeds 4 WMMAs. Scatter-agg via
// native global_atomic_add_f32. Branch-free tanh-GELU (1 exp + 1 rcp).
// ---------------------------------------------------------------------------

typedef __bf16 v16bf __attribute__((ext_vector_type(16)));
typedef float  v8f   __attribute__((ext_vector_type(8)));

#define D_IN 16
#define HDIM 128
#define D_OUT 8
#define MT   4              // M-tiles (16 rows each) per block in edge/node kernels

// ---- WMMA fragment index helpers (CDNA5 ISA 7.12.2, 16-bit A 16x32) --------
__device__ __host__ inline int frag_k_of(int lane, int e) {
    int kgrp = (lane & 16) ? 8 : 0;
    int v = e >> 1, half = e & 1;
    return (v < 4) ? (kgrp + 2 * v + half)
                   : (16 + kgrp + 2 * (v - 4) + half);
}

// Load A fragment (16 rows x 32 K) from row-major bf16 LDS tile.
// Per lane this is two contiguous 16B runs -> compiler emits 2x ds_load_b128.
__device__ inline v16bf load_a_frag(const __bf16* A, int ldk, int kt, int lane) {
    union { v16bf v; __bf16 e[16]; } f;
    int m = lane & 15;
    #pragma unroll
    for (int e = 0; e < 16; ++e) {
        int k = kt * 32 + frag_k_of(lane, e);
        f.e[e] = A[m * ldk + k];
    }
    return f.v;
}

// Load B fragment from pre-packed weights: layout [nt][kt][lane][16] bf16.
__device__ inline v16bf load_b_frag(const __bf16* Wp, int KT, int nt, int kt, int lane) {
    const v16bf* p = (const v16bf*)Wp;
    return p[(size_t)(nt * KT + kt) * 32 + lane];
}

// Multi-M GEMM: one B fragment load feeds MT_ WMMAs (B register reuse).
template <int MT_, int KT>
__device__ inline void wmma_multi(const __bf16* Alds, int ldk, const __bf16* Wp,
                                  int nt, int lane, v8f* acc) {
    #pragma unroll
    for (int kt = 0; kt < KT; ++kt) {
        v16bf b = load_b_frag(Wp, KT, nt, kt, lane);
        #pragma unroll
        for (int mt = 0; mt < MT_; ++mt) {
            v16bf a = load_a_frag(Alds + mt * 16 * ldk, ldk, kt, lane);
            acc[mt] = __builtin_amdgcn_wmma_f32_16x16x32_bf16(
                false, a, false, b, (short)0, acc[mt], false, false);
        }
    }
}

// Branch-free tanh-form GELU: |err| ~1e-3 abs, below bf16 GEMM noise floor.
// tanh(u) = 1 - 2/(1+exp(2u)); saturates correctly at both ends.
__device__ inline float gelu_fast(float x) {
    float u = 0.7978845608028654f * (x + 0.044715f * x * x * x);
    float e = __expf(2.0f * u);
    float t = 1.0f - 2.0f * __builtin_amdgcn_rcpf(1.0f + e);
    return 0.5f * x * (1.0f + t);
}

// ---- weight packing: f32 [K,N] row-major -> bf16 B-fragment layout ---------
__global__ void pack_weights(const float* __restrict__ W, __bf16* __restrict__ out,
                             int K, int N, int KT, int NT) {
    int idx = blockIdx.x * 256 + threadIdx.x;
    int total = NT * KT * 32 * 16;
    if (idx >= total) return;
    int e    = idx & 15;
    int lane = (idx >> 4) & 31;
    int kt   = (idx >> 9) % KT;
    int nt   = (idx >> 9) / KT;
    int n = nt * 16 + (lane & 15);
    int k = kt * 32 + frag_k_of(lane, e);
    __bf16 v = (__bf16)0.0f;
    if (k < K && n < N) v = (__bf16)W[(size_t)k * N + n];
    out[idx] = v;
}

__global__ void zero_f32v4(float4* __restrict__ p, int n4) {
    int i = blockIdx.x * 256 + threadIdx.x;
    if (i < n4) p[i] = make_float4(0.f, 0.f, 0.f, 0.f);
}

// ---- encoder: x[N,16] -> Linear(16,128) -> GELU -> Linear(128,128) ---------
__global__ __launch_bounds__(256)
void encoder_kernel(const float* __restrict__ x,
                    const __bf16* __restrict__ W1p, const float* __restrict__ b1,
                    const __bf16* __restrict__ W2p, const float* __restrict__ b2,
                    float* __restrict__ h, __bf16* __restrict__ hbf, int N) {
    __shared__ __bf16 Ain[16 * 32];    // K padded 16 -> 32
    __shared__ __bf16 Hid[16 * HDIM];
    int t = threadIdx.x, tile = blockIdx.x;
    for (int i = t; i < 16 * 32; i += 256) {
        int m = i >> 5, c = i & 31;
        int node = tile * 16 + m;
        __bf16 v = (__bf16)0.0f;
        if (node < N && c < D_IN) v = (__bf16)x[(size_t)node * D_IN + c];
        Ain[i] = v;
    }
    __syncthreads();
    int lane = t & 31, wave = t >> 5;
    int n = wave * 16 + (lane & 15);
    int mbase = (lane & 16) ? 8 : 0;

    v8f h1[1] = {};
    wmma_multi<1, 1>(Ain, 32, W1p, wave, lane, h1);
    float bb1 = b1[n];
    #pragma unroll
    for (int v = 0; v < 8; ++v)
        Hid[(v + mbase) * HDIM + n] = (__bf16)gelu_fast(h1[0][v] + bb1);
    __syncthreads();

    v8f h2[1] = {};
    wmma_multi<1, 4>(Hid, HDIM, W2p, wave, lane, h2);
    float bb2 = b2[n];
    #pragma unroll
    for (int v = 0; v < 8; ++v) {
        int node = tile * 16 + v + mbase;
        if (node < N) {
            float val = h2[0][v] + bb2;
            h[(size_t)node * HDIM + n]   = val;
            hbf[(size_t)node * HDIM + n] = (__bf16)val;
        }
    }
}

// ---- edge MLP + scatter-add aggregation (64 edges / block) -----------------
__global__ __launch_bounds__(256)
void edge_mlp_kernel(const __bf16* __restrict__ hbf, const int* __restrict__ ei,
                     const __bf16* __restrict__ W1p, const float* __restrict__ b1,
                     const __bf16* __restrict__ W2p, const float* __restrict__ b2,
                     float* __restrict__ agg, int E) {
    __shared__ __bf16 Ain[MT * 16 * 256];   // [h[src] | h[dst]] per edge row
    __shared__ __bf16 Hid[MT * 16 * HDIM];
    __shared__ int sSrc[MT * 16], sDst[MT * 16];
    int t = threadIdx.x;
    int row0 = blockIdx.x * (MT * 16);
    if (t < MT * 16) {
        int e = row0 + t;
        int s = 0, d = 0;
        if (e < E) { s = ei[e]; d = ei[E + e]; }
        sSrc[t] = s; sDst[t] = d;
    }
    __syncthreads();
    // vectorized gather: 8B (4 bf16) per lane; each wave streams a full row
    for (int i = t; i < MT * 16 * 64; i += 256) {
        int m = i >> 6, cg = i & 63;                 // cg = 4-col group
        int node = (cg < 32) ? sSrc[m] : sDst[m];
        ((uint2*)Ain)[i] = ((const uint2*)(hbf + (size_t)node * HDIM))[cg & 31];
    }
    __syncthreads();

    int lane = t & 31, wave = t >> 5;
    int n = wave * 16 + (lane & 15);
    int mbase = (lane & 16) ? 8 : 0;

    v8f acc[MT] = {};
    wmma_multi<MT, 8>(Ain, 256, W1p, wave, lane, acc);
    float bb1 = b1[n];
    #pragma unroll
    for (int mt = 0; mt < MT; ++mt)
        #pragma unroll
        for (int v = 0; v < 8; ++v)
            Hid[(mt * 16 + v + mbase) * HDIM + n] = (__bf16)gelu_fast(acc[mt][v] + bb1);
    __syncthreads();

    v8f msg[MT] = {};
    wmma_multi<MT, 4>(Hid, HDIM, W2p, wave, lane, msg);
    float bb2 = b2[n];
    #pragma unroll
    for (int mt = 0; mt < MT; ++mt) {
        #pragma unroll
        for (int v = 0; v < 8; ++v) {
            int m = mt * 16 + v + mbase;
            int e = row0 + m;
            if (e < E)
                unsafeAtomicAdd(&agg[(size_t)sDst[m] * HDIM + n], msg[mt][v] + bb2);
        }
    }
}

// ---- node MLP + residual (64 nodes / block) --------------------------------
__global__ __launch_bounds__(256)
void node_mlp_kernel(float* __restrict__ h, __bf16* __restrict__ hbf,
                     const float* __restrict__ agg,
                     const __bf16* __restrict__ W1p, const float* __restrict__ b1,
                     const __bf16* __restrict__ W2p, const float* __restrict__ b2,
                     int N) {
    __shared__ __bf16 Ain[MT * 16 * 256];   // [h | agg]
    __shared__ __bf16 Hid[MT * 16 * HDIM];
    int t = threadIdx.x;
    int row0 = blockIdx.x * (MT * 16);
    for (int i = t; i < MT * 16 * 64; i += 256) {
        int m = i >> 6, cg = i & 63;
        int node = row0 + m;
        union { uint2 u; __bf16 e[4]; } val;
        val.u = make_uint2(0u, 0u);
        if (node < N) {
            if (cg < 32) {
                val.u = ((const uint2*)(hbf + (size_t)node * HDIM))[cg];
            } else {
                float4 a4 = ((const float4*)(agg + (size_t)node * HDIM))[cg - 32];
                val.e[0] = (__bf16)a4.x; val.e[1] = (__bf16)a4.y;
                val.e[2] = (__bf16)a4.z; val.e[3] = (__bf16)a4.w;
            }
        }
        ((uint2*)Ain)[i] = val.u;
    }
    __syncthreads();

    int lane = t & 31, wave = t >> 5;
    int n = wave * 16 + (lane & 15);
    int mbase = (lane & 16) ? 8 : 0;

    v8f acc[MT] = {};
    wmma_multi<MT, 8>(Ain, 256, W1p, wave, lane, acc);
    float bb1 = b1[n];
    #pragma unroll
    for (int mt = 0; mt < MT; ++mt)
        #pragma unroll
        for (int v = 0; v < 8; ++v)
            Hid[(mt * 16 + v + mbase) * HDIM + n] = (__bf16)gelu_fast(acc[mt][v] + bb1);
    __syncthreads();

    v8f up[MT] = {};
    wmma_multi<MT, 4>(Hid, HDIM, W2p, wave, lane, up);
    float bb2 = b2[n];
    #pragma unroll
    for (int mt = 0; mt < MT; ++mt) {
        #pragma unroll
        for (int v = 0; v < 8; ++v) {
            int node = row0 + mt * 16 + v + mbase;
            if (node < N) {
                float nh = h[(size_t)node * HDIM + n] + (up[mt][v] + bb2);
                h[(size_t)node * HDIM + n]   = nh;
                hbf[(size_t)node * HDIM + n] = (__bf16)nh;
            }
        }
    }
}

// ---- decoder: Linear(128,128) -> GELU -> Linear(128,8) ---------------------
__global__ __launch_bounds__(256)
void decoder_kernel(const __bf16* __restrict__ hbf,
                    const __bf16* __restrict__ W1p, const float* __restrict__ b1,
                    const __bf16* __restrict__ W2p, const float* __restrict__ b2,
                    float* __restrict__ out, int N) {
    __shared__ __bf16 Ain[16 * HDIM];
    __shared__ __bf16 Hid[16 * HDIM];
    int t = threadIdx.x, tile = blockIdx.x;
    for (int i = t; i < 16 * HDIM; i += 256) {
        int m = i >> 7, c = i & (HDIM - 1);
        int node = tile * 16 + m;
        Ain[i] = (node < N) ? hbf[(size_t)node * HDIM + c] : (__bf16)0.0f;
    }
    __syncthreads();

    int lane = t & 31, wave = t >> 5;
    int n = wave * 16 + (lane & 15);
    int mbase = (lane & 16) ? 8 : 0;

    v8f h1[1] = {};
    wmma_multi<1, 4>(Ain, HDIM, W1p, wave, lane, h1);
    float bb1 = b1[n];
    #pragma unroll
    for (int v = 0; v < 8; ++v)
        Hid[(v + mbase) * HDIM + n] = (__bf16)gelu_fast(h1[0][v] + bb1);
    __syncthreads();

    if (wave == 0) {                       // wave-uniform branch: EXEC stays full
        v8f o[1] = {};
        wmma_multi<1, 4>(Hid, HDIM, W2p, 0, lane, o);
        int nn = lane & 15;
        float bb2 = (nn < D_OUT) ? b2[nn] : 0.0f;
        #pragma unroll
        for (int v = 0; v < 8; ++v) {
            int node = tile * 16 + v + mbase;
            if (node < N && nn < D_OUT)
                out[(size_t)node * D_OUT + nn] = o[0][v] + bb2;
        }
    }
}

// ---------------------------------------------------------------------------
extern "C" void kernel_launch(void* const* d_in, const int* in_sizes, int n_in,
                              void* d_out, int out_size, void* d_ws, size_t ws_size,
                              hipStream_t stream) {
    const float* node_features = (const float*)d_in[0];
    const int*   ei            = (const int*)d_in[1];
    const float* enc_W1 = (const float*)d_in[2];
    const float* enc_b1 = (const float*)d_in[3];
    const float* enc_W2 = (const float*)d_in[4];
    const float* enc_b2 = (const float*)d_in[5];
    const float* eW1 = (const float*)d_in[6];
    const float* eb1 = (const float*)d_in[7];
    const float* eW2 = (const float*)d_in[8];
    const float* eb2 = (const float*)d_in[9];
    const float* nW1 = (const float*)d_in[10];
    const float* nb1 = (const float*)d_in[11];
    const float* nW2 = (const float*)d_in[12];
    const float* nb2 = (const float*)d_in[13];
    const float* dec_W1 = (const float*)d_in[14];
    const float* dec_b1 = (const float*)d_in[15];
    const float* dec_W2 = (const float*)d_in[16];
    const float* dec_b2 = (const float*)d_in[17];

    const int N = in_sizes[0] / D_IN;
    const int E = in_sizes[1] / 2;
    const int L = in_sizes[6] / (2 * HDIM * HDIM);   // N_LAYERS

    // ---- workspace carve-up (256B aligned) ----
    char* ws = (char*)d_ws;
    size_t off = 0;
    auto alloc = [&](size_t bytes) -> void* {
        void* p = ws + off;
        off += (bytes + 255) & ~(size_t)255;
        return p;
    };
    float*  h    = (float*)alloc((size_t)N * HDIM * 4);
    __bf16* hbf  = (__bf16*)alloc((size_t)N * HDIM * 2);
    float*  agg  = (float*)alloc((size_t)N * HDIM * 4);
    const int ELW1 = 8 * 8 * 512;   // 2H x H packed elems
    const int ELW2 = 8 * 4 * 512;   // H  x H packed elems
    __bf16* encW1p = (__bf16*)alloc((size_t)8 * 1 * 512 * 2);
    __bf16* encW2p = (__bf16*)alloc((size_t)ELW2 * 2);
    __bf16* eW1p   = (__bf16*)alloc((size_t)L * ELW1 * 2);
    __bf16* eW2p   = (__bf16*)alloc((size_t)L * ELW2 * 2);
    __bf16* nW1p   = (__bf16*)alloc((size_t)L * ELW1 * 2);
    __bf16* nW2p   = (__bf16*)alloc((size_t)L * ELW2 * 2);
    __bf16* decW1p = (__bf16*)alloc((size_t)ELW2 * 2);
    __bf16* decW2p = (__bf16*)alloc((size_t)1 * 4 * 512 * 2);
    (void)ws_size; (void)n_in; (void)out_size;

    auto pack = [&](const float* W, __bf16* dst, int K, int Nc, int KT, int NT) {
        int total = NT * KT * 32 * 16;
        pack_weights<<<(total + 255) / 256, 256, 0, stream>>>(W, dst, K, Nc, KT, NT);
    };

    pack(enc_W1, encW1p, D_IN, HDIM, 1, 8);
    pack(enc_W2, encW2p, HDIM, HDIM, 4, 8);
    for (int l = 0; l < L; ++l) {
        pack(eW1 + (size_t)l * 2 * HDIM * HDIM, eW1p + (size_t)l * ELW1, 2 * HDIM, HDIM, 8, 8);
        pack(eW2 + (size_t)l * HDIM * HDIM,     eW2p + (size_t)l * ELW2, HDIM,     HDIM, 4, 8);
        pack(nW1 + (size_t)l * 2 * HDIM * HDIM, nW1p + (size_t)l * ELW1, 2 * HDIM, HDIM, 8, 8);
        pack(nW2 + (size_t)l * HDIM * HDIM,     nW2p + (size_t)l * ELW2, HDIM,     HDIM, 4, 8);
    }
    pack(dec_W1, decW1p, HDIM, HDIM, 4, 8);
    pack(dec_W2, decW2p, HDIM, D_OUT, 4, 1);

    const int nodeTiles16 = (N + 15) / 16;
    const int nodeBlocks  = (N + MT * 16 - 1) / (MT * 16);
    const int edgeBlocks  = (E + MT * 16 - 1) / (MT * 16);

    encoder_kernel<<<nodeTiles16, 256, 0, stream>>>(node_features, encW1p, enc_b1,
                                                    encW2p, enc_b2, h, hbf, N);
    for (int l = 0; l < L; ++l) {
        int n4 = (N * HDIM) / 4;
        zero_f32v4<<<(n4 + 255) / 256, 256, 0, stream>>>((float4*)agg, n4);
        edge_mlp_kernel<<<edgeBlocks, 256, 0, stream>>>(
            hbf, ei,
            eW1p + (size_t)l * ELW1, eb1 + (size_t)l * HDIM,
            eW2p + (size_t)l * ELW2, eb2 + (size_t)l * HDIM,
            agg, E);
        node_mlp_kernel<<<nodeBlocks, 256, 0, stream>>>(
            h, hbf, agg,
            nW1p + (size_t)l * ELW1, nb1 + (size_t)l * HDIM,
            nW2p + (size_t)l * ELW2, nb2 + (size_t)l * HDIM,
            N);
    }
    decoder_kernel<<<nodeTiles16, 256, 0, stream>>>(hbf, decW1p, dec_b1,
                                                    decW2p, dec_b2, (float*)d_out, N);
}
